// SelfAttentionWithRelations_18124761989450
// MI455X (gfx1250) — compile-verified, matrix operand order checked by reading, not run on hardware
//
#include <hip/hip_runtime.h>
#include <hip/hip_bf16.h>

typedef __attribute__((ext_vector_type(16))) _Float16 v16h;
typedef __attribute__((ext_vector_type(8)))  float    v8f;

constexpr int Bn = 4;
constexpr int Tn = 256;
constexpr int En = 512;
constexpr int Hn = 8;
constexpr int Sn = 64;
constexpr int BT = Bn * Tn;   // 1024

// ---------------------------------------------------------------------------
// Kernel 0: convert weights to f16 once (Wq/Wk/Wv 64x64, Wu 512x512)
// ---------------------------------------------------------------------------
__global__ __launch_bounds__(256) void prep_weights(
    const float* __restrict__ Wq, const float* __restrict__ Wk,
    const float* __restrict__ Wv, const float* __restrict__ Wu,
    _Float16* __restrict__ wq16, _Float16* __restrict__ wk16,
    _Float16* __restrict__ wv16, _Float16* __restrict__ wu16) {
  int idx = blockIdx.x * 256 + threadIdx.x;
  if (idx < Sn * Sn) {
    wq16[idx] = (_Float16)Wq[idx];
    wk16[idx] = (_Float16)Wk[idx];
    wv16[idx] = (_Float16)Wv[idx];
  }
  if (idx < En * En) wu16[idx] = (_Float16)Wu[idx];
}

// ---------------------------------------------------------------------------
// Kernel 1: QKV projections via WMMA f16 -> f32 accum.
//   Q_h = X_h @ Wq^T  (per head, M=1024, N=64, K=64), scale folded into q.
// One wave computes a 16(M) x 64(N) strip for one (head, {q,k,v}) choice.
// Wave tasks = 64 Mtiles * 8 heads * 3 = 1536 waves -> 192 blocks of 8 waves.
// ---------------------------------------------------------------------------
__global__ __launch_bounds__(256) void qkv_proj(
    const float* __restrict__ x,
    const _Float16* __restrict__ wq16, const _Float16* __restrict__ wk16,
    const _Float16* __restrict__ wv16,
    float* __restrict__ q, float* __restrict__ k, float* __restrict__ v) {
  int wave = (blockIdx.x * blockDim.x + threadIdx.x) >> 5;
  int lane = threadIdx.x & 31;
  int which = wave / (64 * Hn);          // 0=q 1=k 2=v
  int rem   = wave % (64 * Hn);
  int mt    = rem / Hn;                  // M tile (16 rows of b*t)
  int h     = rem % Hn;
  const _Float16* W = (which == 0) ? wq16 : (which == 1) ? wk16 : wv16;
  float* out        = (which == 0) ? q    : (which == 1) ? k    : v;
  const float sc    = (which == 0) ? 0.125f : 1.0f;  // 1/sqrt(64) folded into q

  int m0 = mt * 16;
  int g  = lane >> 4;        // K-half select (ISA A/B layout)
  int mr = lane & 15;        // A row / B column / D column

  for (int nt = 0; nt < 4; ++nt) {
    v8f acc = {};
    const _Float16* brow = W + (size_t)(nt * 16 + mr) * Sn;  // B[k][n] = W[n][k]
    const float*    arow = x + (size_t)(m0 + mr) * En + h * Sn;
    for (int kt = 0; kt < 2; ++kt) {
      v16h A, Bm;
#pragma unroll
      for (int vv = 0; vv < 8; ++vv) {
        int kb = kt * 32 + 2 * vv + ((vv >= 4) ? 8 : 0) + 8 * g;
        A[2 * vv]     = (_Float16)arow[kb];
        A[2 * vv + 1] = (_Float16)arow[kb + 1];
        Bm[2 * vv]     = brow[kb];
        Bm[2 * vv + 1] = brow[kb + 1];
      }
      acc = __builtin_amdgcn_wmma_f32_16x16x32_f16(false, A, false, Bm,
                                                   (short)0, acc, false, false);
    }
#pragma unroll
    for (int r = 0; r < 8; ++r) {
      int m   = m0 + r + 8 * g;
      int col = h * Sn + nt * 16 + mr;
      out[(size_t)m * En + col] = acc[r] * sc;
    }
  }
}

// ---------------------------------------------------------------------------
// Kernel 2: streaming relation attention. One block per (b,i); 8 waves.
// Phase 1: score[j][h] = q_h . (k[j]_h + ek[i,j]_h)      (q pre-scaled)
// Phase 2: masked softmax over j (one wave per head)
// Phase 3: vec[i]_h = sum_j prob[j][h] * (v[j]_h + ev[i,j]_h)
// This streams 1 MB of relation data per block; the kernel is HBM-bound.
// ---------------------------------------------------------------------------
__global__ __launch_bounds__(256) void attn_stream(
    const float* __restrict__ q,  const float* __restrict__ kk,
    const float* __restrict__ vv, const float* __restrict__ relk,
    const float* __restrict__ relv, const int* __restrict__ mask,
    float* __restrict__ vec) {
  __shared__ float s_score[Tn * Hn];     // [j][h], 8 KB
  const int bi   = blockIdx.x;           // b*T + i
  const int b    = bi >> 8;
  const int tid  = threadIdx.x;
  const int wave = tid >> 5, lane = tid & 31;

  // ---- Phase 1: lane owns e-range [16*lane, 16*lane+16) -> head = lane>>2
  float qseg[16];
  {
    const float* qrow = q + (size_t)bi * En + 16 * lane;
#pragma unroll
    for (int t2 = 0; t2 < 4; ++t2) {
      float4 f = *(const float4*)(qrow + 4 * t2);
      qseg[4 * t2 + 0] = f.x; qseg[4 * t2 + 1] = f.y;
      qseg[4 * t2 + 2] = f.z; qseg[4 * t2 + 3] = f.w;
    }
  }
  const float* ekbase = relk + (size_t)bi * Tn * En;
  for (int j = wave; j < Tn; j += 8) {
    const float* ekrow = ekbase + (size_t)j * En + 16 * lane;
    const float* krow  = kk + (size_t)(b * Tn + j) * En + 16 * lane;
    __builtin_prefetch(ekrow + 8 * En, 0, 1);   // global_prefetch_b8, 8 rows ahead
    float p = 0.f;
#pragma unroll
    for (int t2 = 0; t2 < 4; ++t2) {
      float4 a = *(const float4*)(ekrow + 4 * t2);
      float4 c = *(const float4*)(krow + 4 * t2);
      p += (a.x + c.x) * qseg[4 * t2 + 0] + (a.y + c.y) * qseg[4 * t2 + 1] +
           (a.z + c.z) * qseg[4 * t2 + 2] + (a.w + c.w) * qseg[4 * t2 + 3];
    }
    // reduce the 4 lanes covering one head segment (wave32 shuffles)
    p += __shfl_xor(p, 1, 32);
    p += __shfl_xor(p, 2, 32);
    if ((lane & 3) == 0) s_score[j * Hn + (lane >> 2)] = p;
  }
  __syncthreads();

  // ---- Phase 2: softmax over j; wave w owns head h=w
  {
    const int h = wave;
    const int* mrow = mask + (size_t)bi * Tn;
    float vals[8];
    float mx = -3.0e38f;
#pragma unroll
    for (int r = 0; r < 8; ++r) {
      int j = lane + 32 * r;
      float s = s_score[j * Hn + h];
      if (mrow[j] == 0) s = -1.0e9f;
      vals[r] = s;
      mx = fmaxf(mx, s);
    }
    for (int off = 1; off < 32; off <<= 1) mx = fmaxf(mx, __shfl_xor(mx, off, 32));
    float sum = 0.f;
#pragma unroll
    for (int r = 0; r < 8; ++r) { vals[r] = __expf(vals[r] - mx); sum += vals[r]; }
    for (int off = 1; off < 32; off <<= 1) sum += __shfl_xor(sum, off, 32);
    float inv = 1.0f / sum;
#pragma unroll
    for (int r = 0; r < 8; ++r) s_score[(lane + 32 * r) * Hn + h] = vals[r] * inv;
  }
  __syncthreads();

  // ---- Phase 3: wave w owns head h=w; lane owns 2 floats of that head
  {
    const int h  = wave;
    const int e0 = h * Sn + 2 * lane;
    float2 acc = {0.f, 0.f};
    const float* evbase = relv + (size_t)bi * Tn * En + e0;
    const float* vbase  = vv + (size_t)(b * Tn) * En + e0;
    for (int j = 0; j < Tn; ++j) {
      __builtin_prefetch(evbase + (size_t)(j + 8) * En, 0, 1);
      float  pr  = s_score[j * Hn + h];     // LDS broadcast (same addr per wave)
      float2 ev2 = *(const float2*)(evbase + (size_t)j * En);
      float2 v2  = *(const float2*)(vbase + (size_t)j * En);
      acc.x += pr * (ev2.x + v2.x);
      acc.y += pr * (ev2.y + v2.y);
    }
    *(float2*)(vec + (size_t)bi * En + e0) = acc;
  }
}

// ---------------------------------------------------------------------------
// Kernel 3: unify-heads GEMM: out = vec @ Wu^T + bu  (M=1024, N=512, K=512)
// WMMA f16 inputs, f32 accumulation. One wave per 16x16 tile, 2048 tiles.
// ---------------------------------------------------------------------------
__global__ __launch_bounds__(256) void unify_gemm(
    const float* __restrict__ vec, const _Float16* __restrict__ wu16,
    const float* __restrict__ bu, float* __restrict__ out) {
  int wave = (blockIdx.x * blockDim.x + threadIdx.x) >> 5;
  int lane = threadIdx.x & 31;
  int mt = wave >> 5;          // 64 M tiles
  int nt = wave & 31;          // 32 N tiles
  int g  = lane >> 4;
  int mr = lane & 15;
  int m0 = mt * 16, n0 = nt * 16;

  v8f acc = {};
  const float*    arow = vec  + (size_t)(m0 + mr) * En;
  const _Float16* brow = wu16 + (size_t)(n0 + mr) * En;   // B[k][n] = Wu[n][k]
  for (int kt = 0; kt < 16; ++kt) {
    v16h A, Bm;
#pragma unroll
    for (int vv2 = 0; vv2 < 8; ++vv2) {
      int kb = kt * 32 + 2 * vv2 + ((vv2 >= 4) ? 8 : 0) + 8 * g;
      A[2 * vv2]     = (_Float16)arow[kb];
      A[2 * vv2 + 1] = (_Float16)arow[kb + 1];
      Bm[2 * vv2]     = brow[kb];
      Bm[2 * vv2 + 1] = brow[kb + 1];
    }
    acc = __builtin_amdgcn_wmma_f32_16x16x32_f16(false, A, false, Bm,
                                                 (short)0, acc, false, false);
  }
  int n = n0 + mr;
  float bias = bu[n];
#pragma unroll
  for (int r = 0; r < 8; ++r)
    out[(size_t)(m0 + r + 8 * g) * En + n] = acc[r] + bias;
}

// ---------------------------------------------------------------------------
// Host launcher
// ---------------------------------------------------------------------------
extern "C" void kernel_launch(void* const* d_in, const int* in_sizes, int n_in,
                              void* d_out, int out_size, void* d_ws, size_t ws_size,
                              hipStream_t stream) {
  (void)in_sizes; (void)n_in; (void)out_size; (void)ws_size;
  const float* x    = (const float*)d_in[0];
  const float* relk = (const float*)d_in[1];
  const float* relv = (const float*)d_in[2];
  const int*   mask = (const int*)d_in[3];
  const float* Wq   = (const float*)d_in[4];
  const float* Wk   = (const float*)d_in[5];
  const float* Wv   = (const float*)d_in[6];
  const float* Wu   = (const float*)d_in[7];
  const float* bu   = (const float*)d_in[8];
  float* out = (float*)d_out;

  // Workspace layout (floats): q | k | v | vec | f16 weights
  float* ws  = (float*)d_ws;
  float* q   = ws;
  float* k   = q + (size_t)BT * En;
  float* v   = k + (size_t)BT * En;
  float* vec = v + (size_t)BT * En;
  _Float16* wq16 = (_Float16*)(vec + (size_t)BT * En);
  _Float16* wk16 = wq16 + Sn * Sn;
  _Float16* wv16 = wk16 + Sn * Sn;
  _Float16* wu16 = wv16 + Sn * Sn;

  prep_weights<<<(En * En + 255) / 256, 256, 0, stream>>>(
      Wq, Wk, Wv, Wu, wq16, wk16, wv16, wu16);
  qkv_proj<<<192, 256, 0, stream>>>(x, wq16, wk16, wv16, q, k, v);
  attn_stream<<<BT, 256, 0, stream>>>(q, k, v, relk, relv, mask, vec);
  unify_gemm<<<256, 256, 0, stream>>>(vec, wu16, bu, out);
}
